// HOPEBlock_13898514170299
// MI455X (gfx1250) — matrix operand order repaired and checked by reading, not compiled, not run on hardware
//
#include <hip/hip_runtime.h>
#include <cstdint>
#include <cstddef>

// ---------------------------------------------------------------------------
// Types
// ---------------------------------------------------------------------------
typedef __attribute__((ext_vector_type(16))) __bf16       v16bf;
typedef __attribute__((ext_vector_type(8)))  float        v8f;
typedef __attribute__((ext_vector_type(4)))  unsigned int u32x4;
typedef __attribute__((ext_vector_type(8)))  int          i32x8;
typedef __attribute__((ext_vector_type(4)))  int          i32x4;

union FragBF { u32x4 u[2]; v16bf v; };

// Native f32<->bf16 conversion (gfx1250 has hardware cvt; RNE)
__device__ __forceinline__ __bf16 f2bf(float f) { return (__bf16)f; }
__device__ __forceinline__ float  bf2f(__bf16 b) { return (float)b; }

__device__ __forceinline__ v8f vzero8() {
  v8f z;
  for (int i = 0; i < 8; ++i) z[i] = 0.f;
  return z;
}
__device__ __forceinline__ v8f wmma_bf16(const FragBF& a, const FragBF& b, v8f c) {
  return __builtin_amdgcn_wmma_f32_16x16x32_bf16(
      /*neg_a=*/false, a.v, /*neg_b=*/false, b.v,
      /*c_mod=*/(short)0, c, /*reuse_a=*/false, /*reuse_b=*/false);
}
__device__ __forceinline__ float gelu_tanh(float v) {
  return 0.5f * v * (1.f + tanhf(0.7978845608028654f * (v + 0.044715f * v * v * v)));
}

// ---------------------------------------------------------------------------
// Problem constants
// ---------------------------------------------------------------------------
#define BB   4
#define SS   2048
#define DIMC 1024
#define HEADSC 16
#define HDC  64
#define MTOT (BB * SS)      // 8192

// ---------------------------------------------------------------------------
// Tensor Data Mover support (gfx1250). Guarded: falls back to manual loads.
// ---------------------------------------------------------------------------
#if defined(__has_builtin)
#if __has_builtin(__builtin_amdgcn_tensor_load_to_lds) && \
    __has_builtin(__builtin_amdgcn_s_wait_tensorcnt)
#define HAVE_TDM 1
#endif
#endif
#ifndef HAVE_TDM
#define HAVE_TDM 0
#endif

#define GBK  32   // GEMM K-tile (one WMMA K-step)
#define GLDT 40   // LDS pitch in bf16: 80B = 64B data + 16B pad (TDM pad emits this)

#if HAVE_TDM
// Issue a TDM 2D tile load: 128 rows x 32 bf16 from a row-major [rows x d0]
// tensor (row pitch = stride elements) into LDS at lds_byte with 16B row pad
// (pad_interval = 16 dwords, pad_amount = 4 dwords -> 80B LDS pitch).
__device__ __forceinline__ void tdm_load_tile(const __bf16* gsrc,
                                              unsigned lds_byte,
                                              unsigned tensor_d0,
                                              unsigned tensor_d1,
                                              unsigned long long stride_elems) {
  const unsigned long long ga = (unsigned long long)(uintptr_t)gsrc;
  u32x4 g0;
  g0[0] = 1u;                                         // count=1, user mode
  g0[1] = lds_byte;                                   // lds_addr
  g0[2] = (unsigned)(ga & 0xffffffffu);               // global_addr[31:0]
  g0[3] = (unsigned)((ga >> 32) & 0x01ffffffu)        // global_addr[56:32]
        | (2u << 30);                                 // type = 2 ("image")
  i32x8 g1;
  g1[0] = (int)((1u << 16)       // data_size = 1 -> 2 bytes
              | (1u << 20)       // pad_enable
              | (3u << 22)       // pad_interval: 2^(3+1) = 16 dwords (64B row)
              | (3u << 25));     // pad_amount: 4 dwords (16B)
  g1[1] = (int)(tensor_d0 << 16);                         // tensor_dim0[15:0]
  g1[2] = (int)((tensor_d0 >> 16) | (tensor_d1 << 16));   // dim0 hi | dim1 lo
  g1[3] = (int)((tensor_d1 >> 16) | ((unsigned)GBK << 16)); // dim1 hi | tile_dim0
  g1[4] = (int)128u;                                      // tile_dim1=128, tile_dim2=0
  g1[5] = (int)(unsigned)(stride_elems & 0xffffffffu);    // dim0_stride lo
  g1[6] = (int)(unsigned)((stride_elems >> 32) & 0xffffu);// dim0_stride hi
  g1[7] = 0;
  const i32x4 z4 = {0, 0, 0, 0};
#if defined(__clang_major__) && (__clang_major__ >= 23)
  const i32x8 z8 = {0, 0, 0, 0, 0, 0, 0, 0};
  __builtin_amdgcn_tensor_load_to_lds(g0, g1, z4, z4, z8, 0);
#else
  __builtin_amdgcn_tensor_load_to_lds(g0, g1, z4, z4, 0);
#endif
}
__device__ __forceinline__ unsigned lds_byte_addr(const void* p) {
  // generic (flat) pointer to LDS: low 32 bits are the LDS byte address
  return (unsigned)(unsigned long long)(uintptr_t)p;
}
#endif

// ---------------------------------------------------------------------------
// Weight transpose + fp32 -> bf16 convert:  Wt[n][k] = W[k][n]
// ---------------------------------------------------------------------------
__global__ __launch_bounds__(256)
void transpose_cvt(const float* __restrict__ W, __bf16* __restrict__ Wt,
                   int K, int N) {
  __shared__ float tile[32][33];
  const int n0 = blockIdx.x * 32;
  const int k0 = blockIdx.y * 32;
  const int tx = threadIdx.x & 31;
  const int ty = threadIdx.x >> 5;   // 0..7
  for (int j = 0; j < 4; ++j)
    tile[ty + j * 8][tx] = W[(size_t)(k0 + ty + j * 8) * N + n0 + tx];
  __syncthreads();
  for (int j = 0; j < 4; ++j)
    Wt[(size_t)(n0 + ty + j * 8) * K + k0 + tx] = f2bf(tile[tx][ty + j * 8]);
}

// ---------------------------------------------------------------------------
// LayerNorm over DIM=1024, fp32 in -> bf16 out. One row per block.
// ---------------------------------------------------------------------------
__global__ __launch_bounds__(256)
void layernorm_bf16(const float* __restrict__ x, const float* __restrict__ g,
                    const float* __restrict__ bta, __bf16* __restrict__ out) {
  __shared__ float rs[256], rs2[256];
  const int row = blockIdx.x;
  const int tid = threadIdx.x;
  float vv[4];
  const float4 v4 = *(const float4*)(x + (size_t)row * DIMC + tid * 4);
  vv[0] = v4.x; vv[1] = v4.y; vv[2] = v4.z; vv[3] = v4.w;
  float s = vv[0] + vv[1] + vv[2] + vv[3];
  float ss = vv[0]*vv[0] + vv[1]*vv[1] + vv[2]*vv[2] + vv[3]*vv[3];
  rs[tid] = s; rs2[tid] = ss;
  __syncthreads();
  for (int off = 128; off > 0; off >>= 1) {
    if (tid < off) { rs[tid] += rs[tid + off]; rs2[tid] += rs2[tid + off]; }
    __syncthreads();
  }
  const float mean = rs[0] * (1.f / DIMC);
  const float var  = rs2[0] * (1.f / DIMC) - mean * mean;
  const float rstd = rsqrtf(var + 1e-5f);
  for (int j = 0; j < 4; ++j) {
    const int i = tid * 4 + j;
    out[(size_t)row * DIMC + i] = f2bf((vv[j] - mean) * rstd * g[i] + bta[i]);
  }
}

// ---------------------------------------------------------------------------
// WMMA bf16 GEMM: out = epilogue(A[M,K] @ Wt[N,K]^T)
//   A row-major bf16, Wt pre-transposed weight [N][K] bf16.
//   BM=BN=128, BK=32; 8 waves, 2x4 WMMA tiles per wave.
//   Tile feed: TDM DMA (double-buffered on TENSORcnt) when available,
//   else manual global->LDS.  Assumes M%128==0, N%128==0, K%32==0.
// ---------------------------------------------------------------------------
enum { EPI_BF16 = 0, EPI_ELU1 = 1, EPI_BIAS_GELU = 2, EPI_BIAS_BF16 = 3,
       EPI_RES_F32 = 4, EPI_BIAS_RES_F32 = 5 };

template <int EPI>
__global__ __launch_bounds__(256)
void gemm_bf16_wmma(const __bf16* __restrict__ A, const __bf16* __restrict__ Wt,
                    const float* __restrict__ bias, const float* __restrict__ res,
                    void* __restrict__ outp, int M, int N, int K) {
  __shared__ alignas(16) __bf16 As[2][128 * GLDT];   // 10KB each
  __shared__ alignas(16) __bf16 Bs[2][128 * GLDT];

  const int tid  = threadIdx.x;
  const int lane = tid & 31;
  const int wave = tid >> 5;
  const int m0 = blockIdx.y * 128;
  const int n0 = blockIdx.x * 128;
  const int wm = (wave >> 1) * 32;   // 0,32,64,96
  const int wn = (wave & 1) * 64;    // 0,64
  const int ml = lane & 15;
  const int hf = lane >> 4;

  v8f acc[2][4];
  for (int i = 0; i < 2; ++i)
    for (int j = 0; j < 4; ++j) acc[i][j] = vzero8();

  const int nIter = K / GBK;

#if HAVE_TDM
  const unsigned la[2] = { lds_byte_addr(&As[0][0]), lds_byte_addr(&As[1][0]) };
  const unsigned lb[2] = { lds_byte_addr(&Bs[0][0]), lds_byte_addr(&Bs[1][0]) };
  if (wave == 0) {   // prologue: DMA first tile pair into buffer 0
    tdm_load_tile(A  + (size_t)m0 * K, la[0], (unsigned)K, 128u, (unsigned long long)K);
    tdm_load_tile(Wt + (size_t)n0 * K, lb[0], (unsigned)K, 128u, (unsigned long long)K);
  }
#endif

  for (int it = 0; it < nIter; ++it) {
    const int cur = it & 1;
#if HAVE_TDM
    if (wave == 0) {
      if (it + 1 < nIter) {   // prefetch next tile pair into the other buffer
        const int nxt = cur ^ 1;
        const int k1 = (it + 1) * GBK;
        tdm_load_tile(A  + (size_t)m0 * K + k1, la[nxt], (unsigned)K, 128u,
                      (unsigned long long)K);
        tdm_load_tile(Wt + (size_t)n0 * K + k1, lb[nxt], (unsigned)K, 128u,
                      (unsigned long long)K);
        __builtin_amdgcn_s_wait_tensorcnt((short)2);  // current pair retired
      } else {
        __builtin_amdgcn_s_wait_tensorcnt((short)0);
      }
    }
    __syncthreads();          // publish buffer `cur` to all waves
#else
    __syncthreads();          // previous compute done reading
    const int k0 = it * GBK;
    // 128 rows x 32 bf16 = 512 chunks of 16B; 2 chunks per thread per matrix
    for (int i = 0; i < 2; ++i) {
      const int c  = tid + i * 256;
      const int r  = c >> 2;
      const int cc = (c & 3) * 8;
      *(u32x4*)(&As[cur][r * GLDT + cc]) =
          *(const u32x4*)(A + (size_t)(m0 + r) * K + k0 + cc);
      *(u32x4*)(&Bs[cur][r * GLDT + cc]) =
          *(const u32x4*)(Wt + (size_t)(n0 + r) * K + k0 + cc);
    }
    __syncthreads();
#endif

    FragBF a[2], b[4];
    for (int tm = 0; tm < 2; ++tm) {
      const __bf16* p = &As[cur][(wm + tm * 16 + ml) * GLDT + hf * 8];
      a[tm].u[0] = *(const u32x4*)p;
      a[tm].u[1] = *(const u32x4*)(p + 16);
    }
    for (int tn = 0; tn < 4; ++tn) {
      const __bf16* p = &Bs[cur][(wn + tn * 16 + ml) * GLDT + hf * 8];
      b[tn].u[0] = *(const u32x4*)p;
      b[tn].u[1] = *(const u32x4*)(p + 16);
    }
    for (int tm = 0; tm < 2; ++tm)
      for (int tn = 0; tn < 4; ++tn)
        acc[tm][tn] = wmma_bf16(a[tm], b[tn], acc[tm][tn]);

#if HAVE_TDM
    __syncthreads();          // all waves done with `cur` before TDM reuses it
#endif
  }

  // Epilogue. D layout: vgpr r -> m_local = hf*8 + r; n_local = lane&15.
  for (int tm = 0; tm < 2; ++tm)
    for (int tn = 0; tn < 4; ++tn)
      for (int r = 0; r < 8; ++r) {
        const int m = m0 + wm + tm * 16 + hf * 8 + r;
        const int n = n0 + wn + tn * 16 + ml;
        const size_t idx = (size_t)m * N + n;
        float v = acc[tm][tn][r];
        if constexpr (EPI == EPI_BF16) {
          ((__bf16*)outp)[idx] = f2bf(v);
        } else if constexpr (EPI == EPI_ELU1) {
          ((__bf16*)outp)[idx] = f2bf(v > 0.f ? v + 1.f : __expf(v)); // elu+1
        } else if constexpr (EPI == EPI_BIAS_GELU) {
          ((__bf16*)outp)[idx] = f2bf(gelu_tanh(v + bias[n]));
        } else if constexpr (EPI == EPI_BIAS_BF16) {
          ((__bf16*)outp)[idx] = f2bf(v + bias[n]);
        } else if constexpr (EPI == EPI_RES_F32) {
          ((float*)outp)[idx] = v + res[idx];
        } else { // EPI_BIAS_RES_F32
          ((float*)outp)[idx] = v + bias[n] + res[idx];
        }
      }
}

// ---------------------------------------------------------------------------
// Chunked causal linear attention (fast-weight memory), C=64 chunk.
// One workgroup per (b,h). State M^T (64x64) in registers + bf16 LDS mirror.
//   num = maskedS@V + Q@M_prev ; den = q.ksum_prev + rowsum(maskedS) + 1e-6
//   M^T += V^T @ K ; ksum += colsum(K)
// ---------------------------------------------------------------------------
#define ASTR 72

__global__ __launch_bounds__(256)
void linattn64(const __bf16* __restrict__ qb, const __bf16* __restrict__ kb,
               const __bf16* __restrict__ vb, __bf16* __restrict__ ob) {
  __shared__ alignas(16) __bf16 Qs [64 * ASTR];   // [t][d]
  __shared__ alignas(16) __bf16 Ks [64 * ASTR];   // [t][d]   (B^T for Q@K^T)
  __shared__ alignas(16) __bf16 KsT[64 * ASTR];   // [d][t]   (B^T for V^T@K)
  __shared__ alignas(16) __bf16 VsT[64 * ASTR];   // [e][t]   (A for V^T@K, B^T for S@V)
  __shared__ alignas(16) __bf16 Ss [64 * ASTR];   // masked S [t][t']
  __shared__ alignas(16) __bf16 MtS[64 * ASTR];   // M^T [e][d]  (B^T for Q@M)
  __shared__ float den[64], ksum[64], ksumN[64];

  const int bh  = blockIdx.x;
  const int b   = bh >> 4;
  const int h   = bh & 15;
  const int tid = threadIdx.x;
  const int lane = tid & 31;
  const int wave = tid >> 5;
  const int ml = lane & 15;
  const int hf = lane >> 4;
  const int tm  = wave >> 1;          // row tile 0..3
  const int tnb = (wave & 1) * 2;     // col tiles {tnb, tnb+1}
  const size_t base = ((size_t)b * SS) * DIMC + (size_t)h * HDC;

  v8f accM[2] = { vzero8(), vzero8() };

  for (int i = tid; i < 64 * ASTR / 2; i += 256) ((unsigned*)MtS)[i] = 0u;
  if (tid < 64) ksum[tid] = 0.f;
  __syncthreads();

  for (int c = 0; c < SS / 64; ++c) {
    const int t0 = c * 64;

    // ---- load Q,K,V chunk; build transposed copies -----------------------
    for (int j = 0; j < 8; ++j) {
      const int p  = tid + j * 256;     // 2048 u32 pairs
      const int t  = p >> 5;
      const int d2 = (p & 31) * 2;
      const size_t g = base + (size_t)(t0 + t) * DIMC + d2;
      const unsigned qw = *(const unsigned*)(qb + g);
      *(unsigned*)(&Qs[t * ASTR + d2]) = qw;
      const unsigned kw = *(const unsigned*)(kb + g);
      *(unsigned*)(&Ks[t * ASTR + d2]) = kw;
      KsT[(d2 + 0) * ASTR + t] = __builtin_bit_cast(__bf16, (unsigned short)(kw & 0xffffu));
      KsT[(d2 + 1) * ASTR + t] = __builtin_bit_cast(__bf16, (unsigned short)(kw >> 16));
      const unsigned vw = *(const unsigned*)(vb + g);
      VsT[(d2 + 0) * ASTR + t] = __builtin_bit_cast(__bf16, (unsigned short)(vw & 0xffffu));
      VsT[(d2 + 1) * ASTR + t] = __builtin_bit_cast(__bf16, (unsigned short)(vw >> 16));
    }
    __syncthreads();

    // ---- S = Q @ K^T ------------------------------------------------------
    v8f sA[2] = { vzero8(), vzero8() };
    for (int ks = 0; ks < 64; ks += 32) {
      FragBF a, b0, b1;
      const __bf16* pa = &Qs[(tm * 16 + ml) * ASTR + ks + hf * 8];
      a.u[0] = *(const u32x4*)pa; a.u[1] = *(const u32x4*)(pa + 16);
      const __bf16* p0 = &Ks[((tnb + 0) * 16 + ml) * ASTR + ks + hf * 8];
      b0.u[0] = *(const u32x4*)p0; b0.u[1] = *(const u32x4*)(p0 + 16);
      const __bf16* p1 = &Ks[((tnb + 1) * 16 + ml) * ASTR + ks + hf * 8];
      b1.u[0] = *(const u32x4*)p1; b1.u[1] = *(const u32x4*)(p1 + 16);
      sA[0] = wmma_bf16(a, b0, sA[0]);
      sA[1] = wmma_bf16(a, b1, sA[1]);
    }
    // causal mask (diagonal included) -> Ss bf16
    for (int i = 0; i < 2; ++i)
      for (int r = 0; r < 8; ++r) {
        const int tr = tm * 16 + hf * 8 + r;
        const int tc = (tnb + i) * 16 + ml;
        Ss[tr * ASTR + tc] = f2bf(tc <= tr ? sA[i][r] : 0.f);
      }
    __syncthreads();

    // ---- den (uses OLD ksum) and new ksum --------------------------------
    if (tid < 64) {
      float sD = 1e-6f;
      for (int d = 0; d < 64; ++d)  sD += bf2f(Qs[tid * ASTR + d]) * ksum[d];
      for (int tp = 0; tp <= tid; ++tp) sD += bf2f(Ss[tid * ASTR + tp]);
      den[tid] = sD;
    } else if (tid < 128) {
      const int d = tid - 64;
      float sK = ksum[d];
      for (int t = 0; t < 64; ++t) sK += bf2f(Ks[t * ASTR + d]);
      ksumN[d] = sK;
    }

    // ---- num = maskedS @ V  +  Q @ M_prev --------------------------------
    v8f nacc[2] = { vzero8(), vzero8() };
    for (int ks = 0; ks < 64; ks += 32) {         // maskedS @ V
      FragBF a, b0, b1;
      const __bf16* pa = &Ss[(tm * 16 + ml) * ASTR + ks + hf * 8];
      a.u[0] = *(const u32x4*)pa; a.u[1] = *(const u32x4*)(pa + 16);
      const __bf16* p0 = &VsT[((tnb + 0) * 16 + ml) * ASTR + ks + hf * 8];
      b0.u[0] = *(const u32x4*)p0; b0.u[1] = *(const u32x4*)(p0 + 16);
      const __bf16* p1 = &VsT[((tnb + 1) * 16 + ml) * ASTR + ks + hf * 8];
      b1.u[0] = *(const u32x4*)p1; b1.u[1] = *(const u32x4*)(p1 + 16);
      nacc[0] = wmma_bf16(a, b0, nacc[0]);
      nacc[1] = wmma_bf16(a, b1, nacc[1]);
    }
    for (int ks = 0; ks < 64; ks += 32) {         // Q @ M_prev
      FragBF a, b0, b1;
      const __bf16* pa = &Qs[(tm * 16 + ml) * ASTR + ks + hf * 8];
      a.u[0] = *(const u32x4*)pa; a.u[1] = *(const u32x4*)(pa + 16);
      const __bf16* p0 = &MtS[((tnb + 0) * 16 + ml) * ASTR + ks + hf * 8];
      b0.u[0] = *(const u32x4*)p0; b0.u[1] = *(const u32x4*)(p0 + 16);
      const __bf16* p1 = &MtS[((tnb + 1) * 16 + ml) * ASTR + ks + hf * 8];
      b1.u[0] = *(const u32x4*)p1; b1.u[1] = *(const u32x4*)(p1 + 16);
      nacc[0] = wmma_bf16(a, b0, nacc[0]);
      nacc[1] = wmma_bf16(a, b1, nacc[1]);
    }
    // ---- M^T += V^T @ K  (kept in registers) -----------------------------
    for (int ks = 0; ks < 64; ks += 32) {
      FragBF a, b0, b1;
      const __bf16* pa = &VsT[(tm * 16 + ml) * ASTR + ks + hf * 8];
      a.u[0] = *(const u32x4*)pa; a.u[1] = *(const u32x4*)(pa + 16);
      const __bf16* p0 = &KsT[((tnb + 0) * 16 + ml) * ASTR + ks + hf * 8];
      b0.u[0] = *(const u32x4*)p0; b0.u[1] = *(const u32x4*)(p0 + 16);
      const __bf16* p1 = &KsT[((tnb + 1) * 16 + ml) * ASTR + ks + hf * 8];
      b1.u[0] = *(const u32x4*)p1; b1.u[1] = *(const u32x4*)(p1 + 16);
      accM[0] = wmma_bf16(a, b0, accM[0]);
      accM[1] = wmma_bf16(a, b1, accM[1]);
    }
    __syncthreads();   // all reads of MtS/ksum/Ss done; den/ksumN written

    // ---- commit state mirror + ksum; emit outputs ------------------------
    for (int i = 0; i < 2; ++i)
      for (int r = 0; r < 8; ++r) {
        const int er = tm * 16 + hf * 8 + r;
        const int dc = (tnb + i) * 16 + ml;
        MtS[er * ASTR + dc] = f2bf(accM[i][r]);
      }
    if (tid < 64) ksum[tid] = ksumN[tid];
    for (int i = 0; i < 2; ++i)
      for (int r = 0; r < 8; ++r) {
        const int tr = tm * 16 + hf * 8 + r;
        const int e  = (tnb + i) * 16 + ml;
        ob[base + (size_t)(t0 + tr) * DIMC + e] = f2bf(nacc[i][r] / den[tr]);
      }
  }
}

// ---------------------------------------------------------------------------
// Host-side orchestration
// ---------------------------------------------------------------------------
extern "C" void kernel_launch(void* const* d_in, const int* in_sizes, int n_in,
                              void* d_out, int out_size, void* d_ws, size_t ws_size,
                              hipStream_t stream) {
  const float* x     = (const float*)d_in[0];
  const float* Wq    = (const float*)d_in[1];
  const float* Wk    = (const float*)d_in[2];
  const float* Wv    = (const float*)d_in[3];
  const float* Wo    = (const float*)d_in[4];
  const float* ln1_g = (const float*)d_in[5];
  const float* ln1_b = (const float*)d_in[6];
  const float* ln2_g = (const float*)d_in[7];
  const float* ln2_b = (const float*)d_in[8];
  const float* cw1   = (const float*)d_in[9];   // [3,1024,4096]
  const float* cb1   = (const float*)d_in[10];  // [3,4096]
  const float* cw2   = (const float*)d_in[11];  // [3,4096,1024]
  const float* cb2   = (const float*)d_in[12];  // [3,1024]
  float* out = (float*)d_out;

  const size_t MB = (size_t)1 << 20;
  char* ws = (char*)d_ws;
  // Workspace layout (200 MB total); later buffers alias dead earlier ones.
  __bf16* WqT = (__bf16*)(ws + 0 * MB);      // [1024][1024]
  __bf16* WkT = (__bf16*)(ws + 2 * MB);
  __bf16* WvT = (__bf16*)(ws + 4 * MB);
  __bf16* WoT = (__bf16*)(ws + 6 * MB);
  __bf16* W1T = (__bf16*)(ws + 8 * MB);      // 3 x [4096][1024]  (24 MB)
  __bf16* W2T = (__bf16*)(ws + 32 * MB);     // 3 x [1024][4096]  (24 MB)
  __bf16* h1  = (__bf16*)(ws + 56 * MB);     // LN1 out; later attn out (ab)
  __bf16* qb  = (__bf16*)(ws + 72 * MB);     // Q; later h2 (LN2 out)
  __bf16* kb  = (__bf16*)(ws + 88 * MB);     // K; later ybuf (CMS inter-level)
  __bf16* vb  = (__bf16*)(ws + 104 * MB);    // V; later tbuf [8192,4096] (64 MB)
  float*  x1  = (float*)(ws + 168 * MB);     // post-attn residual (32 MB)
  __bf16* ab   = h1;
  __bf16* h2   = qb;
  __bf16* ybuf = kb;
  __bf16* tbuf = vb;

  const dim3 blk(256);

  // 1) one-time weight transpose/convert (reused across the whole GEMM set)
  transpose_cvt<<<dim3(32, 32), blk, 0, stream>>>(Wq, WqT, 1024, 1024);
  transpose_cvt<<<dim3(32, 32), blk, 0, stream>>>(Wk, WkT, 1024, 1024);
  transpose_cvt<<<dim3(32, 32), blk, 0, stream>>>(Wv, WvT, 1024, 1024);
  transpose_cvt<<<dim3(32, 32), blk, 0, stream>>>(Wo, WoT, 1024, 1024);
  for (int l = 0; l < 3; ++l) {
    transpose_cvt<<<dim3(128, 32), blk, 0, stream>>>(
        cw1 + (size_t)l * 1024 * 4096, W1T + (size_t)l * 4096 * 1024, 1024, 4096);
    transpose_cvt<<<dim3(32, 128), blk, 0, stream>>>(
        cw2 + (size_t)l * 4096 * 1024, W2T + (size_t)l * 1024 * 4096, 4096, 1024);
  }

  // 2) h1 = LN1(x)
  layernorm_bf16<<<MTOT, blk, 0, stream>>>(x, ln1_g, ln1_b, h1);

  // 3) Q = elu(h1 Wq)+1 ; K = elu(h1 Wk)+1 ; V = h1 Wv
  gemm_bf16_wmma<EPI_ELU1><<<dim3(8, 64), blk, 0, stream>>>(
      h1, WqT, nullptr, nullptr, qb, MTOT, 1024, 1024);
  gemm_bf16_wmma<EPI_ELU1><<<dim3(8, 64), blk, 0, stream>>>(
      h1, WkT, nullptr, nullptr, kb, MTOT, 1024, 1024);
  gemm_bf16_wmma<EPI_BF16><<<dim3(8, 64), blk, 0, stream>>>(
      h1, WvT, nullptr, nullptr, vb, MTOT, 1024, 1024);

  // 4) chunked causal linear attention (writes ab, aliases h1)
  linattn64<<<BB * HEADSC, blk, 0, stream>>>(qb, kb, vb, ab);

  // 5) x1 = x + ab @ Wo
  gemm_bf16_wmma<EPI_RES_F32><<<dim3(8, 64), blk, 0, stream>>>(
      ab, WoT, nullptr, x, x1, MTOT, 1024, 1024);

  // 6) h2 = LN2(x1)
  layernorm_bf16<<<MTOT, blk, 0, stream>>>(x1, ln2_g, ln2_b, h2);

  // 7) CMS: 3 chained MLPs; final level fuses bias + residual -> fp32 d_out
  for (int l = 0; l < 3; ++l) {
    const __bf16* act = (l == 0) ? h2 : ybuf;
    gemm_bf16_wmma<EPI_BIAS_GELU><<<dim3(32, 64), blk, 0, stream>>>(
        act, W1T + (size_t)l * 4096 * 1024, cb1 + (size_t)l * 4096, nullptr,
        tbuf, MTOT, 4096, 1024);
    if (l < 2) {
      gemm_bf16_wmma<EPI_BIAS_BF16><<<dim3(8, 64), blk, 0, stream>>>(
          tbuf, W2T + (size_t)l * 1024 * 4096, cb2 + (size_t)l * 1024, nullptr,
          ybuf, MTOT, 1024, 4096);
    } else {
      gemm_bf16_wmma<EPI_BIAS_RES_F32><<<dim3(8, 64), blk, 0, stream>>>(
          tbuf, W2T + (size_t)l * 1024 * 4096, cb2 + (size_t)l * 1024, x1,
          out, MTOT, 1024, 4096);
    }
  }
}
